// SeqCoreDetector_72679436583225
// MI455X (gfx1250) — matrix operand order repaired and checked by reading, not compiled, not run on hardware
//
#include <hip/hip_runtime.h>
#include <hip/hip_bf16.h>

#define B_SZ   32
#define T_SZ   2048
#define D_IN   1024
#define D_H    64
#define D_C    16
#define N_TAGS 2
#define KW     7
#define KCONV  (D_H * KW)   // 448 = 14 * 32

typedef __attribute__((ext_vector_type(16))) __bf16 v16bf;
typedef __attribute__((ext_vector_type(8)))  float  v8f;

// ---------------------------------------------------------------------------
// Kernel 1: h[b, j, t] = relu( sum_d x[b,t,d] * W1[j,d] + b1[j] )
// Stored channel-major (B, D_H, T) so the conv kernel reads it as NCH.
// Per-wave tile: 64 j (4 A-tiles of W1) x 32 t (2 B-tiles of x^T), 8 WMMA
// accumulators. Each A fragment feeds 2 WMMAs -> halves W1 L2 traffic and
// doubles independent work between dependent WMMAs (fills hazard slots).
// ---------------------------------------------------------------------------
__global__ __launch_bounds__(256) void k_gemm1(const float* __restrict__ x,
                                               const float* __restrict__ W1,
                                               const float* __restrict__ b1,
                                               float* __restrict__ h) {
    const int lane  = threadIdx.x & 31;
    const int wave  = threadIdx.x >> 5;
    const int tile  = blockIdx.x * 8 + wave;     // 0..2047  (B * T/32)
    const int b     = tile >> 6;                 // tile / (T/32)
    const int tbase = (tile & 63) << 5;          // 32-wide t tile
    const int n     = lane & 15;                 // N column within 16-tile
    const int hi    = lane >> 4;                 // lane half
    const int t0    = tbase + n;
    const int t1    = tbase + 16 + n;

    const float* xrow0 = x + ((size_t)b * T_SZ + t0) * D_IN;
    const float* xrow1 = x + ((size_t)b * T_SZ + t1) * D_IN;
    const int koffB  = hi * 8;                   // B frag: K = koffB + i (+16 for i>=8)
    const int kbaseA = hi * 8;                   // A frag: K = kbaseA + i (+16 for i>=8)

    v8f acc[4][2] = {};
    for (int k0 = 0; k0 < D_IN; k0 += 32) {
        // Two B fragments: 8 floats at k0+koffB and 8 at k0+koffB+16 of each lane's x rows -> bf16
        v16bf bf0, bf1;
        const float* xp0 = xrow0 + k0 + koffB;
        const float* xp1 = xrow1 + k0 + koffB;
        #pragma unroll
        for (int i = 0; i < 8; ++i) { bf0[i] = (__bf16)xp0[i]; bf1[i] = (__bf16)xp1[i]; bf0[i + 8] = (__bf16)xp0[i + 16]; bf1[i + 8] = (__bf16)xp1[i + 16]; }

        #pragma unroll
        for (int jt = 0; jt < 4; ++jt) {
            const int j = jt * 16 + n;           // A row m = lane&15
            const float* wp = W1 + (size_t)j * D_IN + k0 + kbaseA;
            v16bf af;
            #pragma unroll
            for (int i = 0; i < 8; ++i) {
                af[i]     = (__bf16)wp[i];
                af[i + 8] = (__bf16)wp[i + 16];
            }
            acc[jt][0] = __builtin_amdgcn_wmma_f32_16x16x32_bf16(
                false, af, false, bf0, (short)0, acc[jt][0], false, false);
            acc[jt][1] = __builtin_amdgcn_wmma_f32_16x16x32_bf16(
                false, af, false, bf1, (short)0, acc[jt][1], false, false);
        }
    }

    // Epilogue: D VGPR r holds (M = r + 8*hi, N = lane&15)
    #pragma unroll
    for (int jt = 0; jt < 4; ++jt) {
        #pragma unroll
        for (int p = 0; p < 2; ++p) {
            const int t = tbase + p * 16 + n;
            #pragma unroll
            for (int r = 0; r < 8; ++r) {
                int j = jt * 16 + r + hi * 8;
                float v = acc[jt][p][r] + b1[j];
                v = v > 0.f ? v : 0.f;
                h[((size_t)b * D_H + j) * T_SZ + t] = v;
            }
        }
    }
}

// ---------------------------------------------------------------------------
// Kernel 2: conv1d (pad 3) + relu + W2 projection -> emis[b, t, n]
// WMMA: A = Wc flattened (16 c-rows x 448 K), B = im2col (448 K x 16 t).
// K index decomposes as j*7 + kk; boundary taps read 0.
// ---------------------------------------------------------------------------
__global__ __launch_bounds__(256) void k_conv_emis(const float* __restrict__ h,
                                                   const float* __restrict__ Wc,
                                                   const float* __restrict__ bc,
                                                   const float* __restrict__ W2,
                                                   const float* __restrict__ b2,
                                                   float* __restrict__ emis) {
    const int lane  = threadIdx.x & 31;
    const int wave  = threadIdx.x >> 5;
    const int tile  = blockIdx.x * 8 + wave;     // 0..4095
    const int b     = tile >> 7;
    const int tbase = (tile & 127) << 4;
    const int n     = lane & 15;
    const int hi    = lane >> 4;
    const int t     = tbase + n;
    const int c     = n;                         // A row
    const int kbaseA = hi * 8;
    const int koffB  = hi * 8;
    const float* hb = h + (size_t)b * D_H * T_SZ;

    v8f acc = {};
    #pragma unroll
    for (int k0 = 0; k0 < KCONV; k0 += 32) {
        v16bf af, bf;
        #pragma unroll
        for (int i = 0; i < 8; ++i) {
            af[i]     = (__bf16)Wc[c * KCONV + k0 + kbaseA + i];
            af[i + 8] = (__bf16)Wc[c * KCONV + k0 + kbaseA + 16 + i];
        }
        #pragma unroll
        for (int i = 0; i < 16; ++i) {
            int kidx = k0 + koffB + (i < 8 ? i : i + 8); // compile-time after unroll
            int j  = kidx / KW;
            int kk = kidx - j * KW;
            int tt = t + kk - 3;
            float v = (tt >= 0 && tt < T_SZ) ? hb[(size_t)j * T_SZ + tt] : 0.f;
            bf[i] = (__bf16)v;
        }
        acc = __builtin_amdgcn_wmma_f32_16x16x32_bf16(
            false, af, false, bf, (short)0, acc, false, false);
    }

    // relu + bc, then fold 16->2 via W2; each lane holds 8 c-channels of col t
    float p0 = 0.f, p1 = 0.f;
    #pragma unroll
    for (int r = 0; r < 8; ++r) {
        int cc = r + hi * 8;
        float v = acc[r] + bc[cc];
        v = v > 0.f ? v : 0.f;
        p0 += v * W2[cc];
        p1 += v * W2[D_C + cc];
    }
    p0 += __shfl_xor(p0, 16, 32);
    p1 += __shfl_xor(p1, 16, 32);
    if (hi == 0) {
        size_t o = ((size_t)b * T_SZ + t) * N_TAGS;
        emis[o]     = p0 + b2[0];
        emis[o + 1] = p1 + b2[1];
    }
}

// ---------------------------------------------------------------------------
// Kernel 3: CRF NLL. One block per batch. 2-tag forward recursion is a 2x2
// log-semiring matrix product (associative) -> each lane composes a 64-step
// transfer matrix, lane 0 folds the 32 chunks. Numerator is a masked sum.
// ---------------------------------------------------------------------------
__device__ __forceinline__ float lse2(float a, float b) {
    float mx = fmaxf(a, b);
    float mn = fminf(a, b);
    return mx + log1pf(__expf(mn - mx));
}

__global__ __launch_bounds__(32) void k_crf(const float* __restrict__ emis,
                                            const unsigned char* __restrict__ mask,
                                            const int* __restrict__ labels,
                                            const float* __restrict__ start_t,
                                            const float* __restrict__ end_t,
                                            const float* __restrict__ trans,
                                            float* __restrict__ partial) {
    const int b    = blockIdx.x;
    const int lane = threadIdx.x;
    const float* em = emis + (size_t)b * T_SZ * N_TAGS;
    const unsigned char* mk = mask + (size_t)b * T_SZ;
    const int* lb = labels + (size_t)b * T_SZ;
    const float t00 = trans[0], t01 = trans[1], t10 = trans[2], t11 = trans[3];

    const float NEG = -1e30f;
    float m00 = 0.f, m01 = NEG, m10 = NEG, m11 = 0.f;   // identity in log semiring
    float numpart = 0.f;
    int   cntpart = 0;

    int ts = 1 + lane * 64;
    int te = ts + 64; if (te > T_SZ) te = T_SZ;
    for (int t = ts; t < te; ++t) {
        float e0 = em[t * 2], e1 = em[t * 2 + 1];
        if (mk[t]) {
            float s00 = t00 + e0, s01 = t01 + e1;
            float s10 = t10 + e0, s11 = t11 + e1;
            float n00 = lse2(m00 + s00, m01 + s10);
            float n01 = lse2(m00 + s01, m01 + s11);
            float n10 = lse2(m10 + s00, m11 + s10);
            float n11 = lse2(m10 + s01, m11 + s11);
            m00 = n00; m01 = n01; m10 = n10; m11 = n11;
            int lp = lb[t - 1], lt = lb[t];
            numpart += trans[lp * 2 + lt] + em[t * 2 + lt];
        }
    }
    for (int t = lane * 64; t < lane * 64 + 64; ++t) cntpart += (mk[t] != 0);

    __shared__ float sm[32][4];
    __shared__ float snum[32];
    __shared__ int   scnt[32];
    sm[lane][0] = m00; sm[lane][1] = m01; sm[lane][2] = m10; sm[lane][3] = m11;
    snum[lane] = numpart; scnt[lane] = cntpart;
    __syncthreads();

    if (lane == 0) {
        float a0 = start_t[0] + em[0];
        float a1 = start_t[1] + em[1];
        for (int cI = 0; cI < 32; ++cI) {
            float c00 = sm[cI][0], c01 = sm[cI][1], c10 = sm[cI][2], c11 = sm[cI][3];
            float na0 = lse2(a0 + c00, a1 + c10);
            float na1 = lse2(a0 + c01, a1 + c11);
            a0 = na0; a1 = na1;
        }
        float logZ = lse2(a0 + end_t[0], a1 + end_t[1]);
        float num = 0.f; int cnt = 0;
        for (int cI = 0; cI < 32; ++cI) { num += snum[cI]; cnt += scnt[cI]; }
        int l0 = lb[0];
        num += start_t[l0] + em[l0];
        num += end_t[lb[cnt - 1]];
        partial[b] = logZ - num;
    }
}

__global__ __launch_bounds__(32) void k_reduce(const float* __restrict__ partial,
                                               float* __restrict__ out) {
    float v = partial[threadIdx.x];
    #pragma unroll
    for (int off = 16; off; off >>= 1) v += __shfl_down(v, off, 32);
    if (threadIdx.x == 0) out[0] = v;   // = -sum(num - logZ)
}

// ---------------------------------------------------------------------------
extern "C" void kernel_launch(void* const* d_in, const int* in_sizes, int n_in,
                              void* d_out, int out_size, void* d_ws, size_t ws_size,
                              hipStream_t stream) {
    const float*         x       = (const float*)d_in[0];
    const unsigned char* mask    = (const unsigned char*)d_in[1];
    const int*           labels  = (const int*)d_in[2];
    const float*         W1      = (const float*)d_in[3];
    const float*         b1      = (const float*)d_in[4];
    const float*         Wc      = (const float*)d_in[5];
    const float*         bc      = (const float*)d_in[6];
    const float*         W2      = (const float*)d_in[7];
    const float*         b2      = (const float*)d_in[8];
    const float*         start_t = (const float*)d_in[9];
    const float*         end_t   = (const float*)d_in[10];
    const float*         trans   = (const float*)d_in[11];

    float* h       = (float*)d_ws;                       // B*D_H*T      = 16 MB
    float* emis    = h + (size_t)B_SZ * D_H * T_SZ;      // B*T*2        = 512 KB
    float* partial = emis + (size_t)B_SZ * T_SZ * N_TAGS;

    const int tiles1  = B_SZ * (T_SZ / 32);              // 2048 wave-tiles (64j x 32t)
    const int blocks1 = tiles1 / 8;                      // 8 waves / block
    const int tiles2  = B_SZ * (T_SZ / 16);              // 4096 wave-tiles
    const int blocks2 = tiles2 / 8;

    k_gemm1    <<<blocks1, 256, 0, stream>>>(x, W1, b1, h);
    k_conv_emis<<<blocks2, 256, 0, stream>>>(h, Wc, bc, W2, b2, emis);
    k_crf      <<<B_SZ, 32, 0, stream>>>(emis, mask, labels, start_t, end_t, trans, partial);
    k_reduce   <<<1, 32, 0, stream>>>(partial, (float*)d_out);
}
